// Transformer_45208825758394
// MI455X (gfx1250) — matrix operand order repaired and checked by reading, not compiled, not run on hardware
//
#include <hip/hip_runtime.h>
#include <stdint.h>

// ---------------- problem constants ----------------
#define V_  32000
#define E_  1024
#define H_  16
#define L_  8
#define T_  1024
#define B_  2
#define HS_ 64
#define M_  (B_ * T_)   // 2048 rows of activations

typedef __attribute__((ext_vector_type(16))) __bf16 v16bf;
typedef __attribute__((ext_vector_type(8)))  float  v8f;
typedef __attribute__((ext_vector_type(4)))  int    v4i;

// -------- CDNA5 async global->LDS copy (ASYNCcnt path), guarded ----------
// Probe result (round 2 stderr): builtin exists; param0 = int4 addrspace(1)*,
// param1 = int4 addrspace(3)*, then two constant ints (offset, cpol).
#ifndef __has_builtin
#define __has_builtin(x) 0
#endif
#if __has_builtin(__builtin_amdgcn_global_load_async_to_lds_b128)
#define HAVE_ASYNC 1
#define ASYNC_COPY16(dst, src)                                        \
  __builtin_amdgcn_global_load_async_to_lds_b128(                     \
      (__attribute__((address_space(1))) v4i*)(src),                  \
      (__attribute__((address_space(3))) v4i*)(dst), 0, 0)
#define WAIT_ASYNC() asm volatile("s_wait_asynccnt 0x0" ::: "memory")
#else
#define HAVE_ASYNC 0
#define WAIT_ASYNC()
#endif

// ---------------- helpers ----------------
static __device__ __forceinline__ uint16_t f32_to_bf16(float f) {
  uint32_t u = __float_as_uint(f);
  u += 0x7fffu + ((u >> 16) & 1u);   // round-to-nearest-even
  return (uint16_t)(u >> 16);
}

static __device__ __forceinline__ v8f wmma_bf16(v16bf a, v16bf b, v8f c) {
  return __builtin_amdgcn_wmma_f32_16x16x32_bf16(false, a, false, b, (short)0, c,
                                                 false, false);
}

// A-fragment (16x32 bf16), tile in LDS row-major [row][k], row stride even.
// Lane l: M = l&15; vgpr j: K = 16*half + 2j
static __device__ __forceinline__ v16bf frag_a(const uint16_t* tile, int stride,
                                               int row0, int k0, int lane) {
  int m = lane & 15, hf = lane >> 4;
  union { v16bf v; uint32_t u[8]; } f;
  const uint16_t* base = tile + (size_t)(row0 + m) * stride + k0;
#pragma unroll
  for (int j = 0; j < 8; ++j) {
    int kk = 16 * hf + 2 * j;
    f.u[j] = *(const uint32_t*)(base + kk);
  }
  return f.v;
}

// B-fragment (32x16 bf16), tile stored TRANSPOSED in LDS as [n][k], stride even.
// Lane l: N = l&15; vgpr j: K = 16*half + 2j (+k0).
static __device__ __forceinline__ v16bf frag_b(const uint16_t* tile, int stride,
                                               int n0, int k0, int lane) {
  int n = lane & 15, hf = lane >> 4;
  union { v16bf v; uint32_t u[8]; } f;
  const uint16_t* base = tile + (size_t)(n0 + n) * stride + k0 + 16 * hf;
#pragma unroll
  for (int j = 0; j < 8; ++j) f.u[j] = *(const uint32_t*)(base + 2 * j);
  return f.v;
}

// ---------------- small kernels ----------------
__global__ void k_f32_to_bf16(const float* __restrict__ in,
                              uint16_t* __restrict__ out, long n) {
  long i = (long)blockIdx.x * blockDim.x + threadIdx.x;
  long st = (long)gridDim.x * blockDim.x;
  for (; i < n; i += st) out[i] = f32_to_bf16(in[i]);
}

__global__ void k_embed(const int* __restrict__ idx, const float* __restrict__ tok,
                        const float* __restrict__ pos, float* __restrict__ x) {
  size_t i = (size_t)blockIdx.x * blockDim.x + threadIdx.x;
  if (i >= (size_t)M_ * E_) return;
  int row = (int)(i / E_), e = (int)(i % E_);
  int t = row % T_;
  x[i] = tok[(size_t)idx[row] * E_ + e] + pos[(size_t)t * E_ + e];
}

__global__ __launch_bounds__(256) void k_layernorm(const float* __restrict__ x,
                                                   const float* __restrict__ g,
                                                   const float* __restrict__ bb,
                                                   uint16_t* __restrict__ out) {
  int row = blockIdx.x;
  const float* xr = x + (size_t)row * E_;
  __shared__ float red[256];
  float s = 0.f, s2 = 0.f;
  for (int i = threadIdx.x; i < E_; i += 256) { float v = xr[i]; s += v; s2 += v * v; }
  red[threadIdx.x] = s; __syncthreads();
  for (int o = 128; o > 0; o >>= 1) { if (threadIdx.x < o) red[threadIdx.x] += red[threadIdx.x + o]; __syncthreads(); }
  float mean = red[0] / E_; __syncthreads();
  red[threadIdx.x] = s2; __syncthreads();
  for (int o = 128; o > 0; o >>= 1) { if (threadIdx.x < o) red[threadIdx.x] += red[threadIdx.x + o]; __syncthreads(); }
  float var = red[0] / E_ - mean * mean;
  float rstd = rsqrtf(var + 1e-5f);
  for (int i = threadIdx.x; i < E_; i += 256) {
    float v = (xr[i] - mean) * rstd * g[i] + bb[i];
    out[(size_t)row * E_ + i] = f32_to_bf16(v);
  }
}

// ---------------- tiled bf16 WMMA GEMM (double-buffered, async A) ----------
// C[M,N] = A_bf16[M,K] @ B_bf16[K,N] (+bias)(+resid)(relu?) -> f32 / bf16
// block: 256 thr (8 waves), tile 128x128; wave tile 64x32; K step 32.
__global__ __launch_bounds__(256)
void k_gemm(const uint16_t* __restrict__ A, const uint16_t* __restrict__ Bw,
            const float* __restrict__ bias, const float* __restrict__ resid,
            float* __restrict__ outF, uint16_t* __restrict__ outB,
            int Mdim, int Ndim, int Kdim, int relu) {
  __shared__ uint16_t As[2][128 * 32];   // [row][k], rows 16B-aligned
  __shared__ uint16_t Bs[2][128 * 34];   // transposed [n][k], pad->34
  const int tid = threadIdx.x, lane = tid & 31, wid = tid >> 5;
  const int wm = (wid >> 2) * 64;
  const int wn = (wid & 3) * 32;
  const int m0 = blockIdx.y * 128, n0 = blockIdx.x * 128;

  v8f acc[4][2];
#pragma unroll
  for (int mt = 0; mt < 4; ++mt)
#pragma unroll
    for (int nt = 0; nt < 2; ++nt)
#pragma unroll
      for (int e = 0; e < 8; ++e) acc[mt][nt][e] = 0.f;

  uint4 breg[2];
#if !HAVE_ASYNC
  uint4 areg[2];
#endif

  auto issue = [&](int k0, int buf) {    // A: async->LDS; B: global->regs
#pragma unroll
    for (int i = 0; i < 2; ++i) {
      int li = tid + 256 * i;
#if HAVE_ASYNC
      ASYNC_COPY16(&As[buf][li * 8],
                   A + (size_t)(m0 + (li >> 2)) * Kdim + k0 + (li & 3) * 8);
#else
      areg[i] = *(const uint4*)(A + (size_t)(m0 + (li >> 2)) * Kdim + k0 + (li & 3) * 8);
#endif
      breg[i] = *(const uint4*)(Bw + (size_t)(k0 + (li >> 4)) * Ndim + n0 + (li & 15) * 8);
    }
  };
  auto commit = [&](int buf) {           // B: regs -> LDS transposed (+A if sync)
#pragma unroll
    for (int i = 0; i < 2; ++i) {
      int li = tid + 256 * i;
#if !HAVE_ASYNC
      ((uint4*)As[buf])[li] = areg[i];
#endif
      int kr = li >> 4, seg = li & 15;
      const uint16_t* pv = (const uint16_t*)&breg[i];
#pragma unroll
      for (int e = 0; e < 8; ++e) Bs[buf][(size_t)(seg * 8 + e) * 34 + kr] = pv[e];
    }
  };
  auto compute = [&](int buf) {
    v16bf bfr[2];
#pragma unroll
    for (int nt = 0; nt < 2; ++nt) bfr[nt] = frag_b(Bs[buf], 34, wn + nt * 16, 0, lane);
#pragma unroll
    for (int mt = 0; mt < 4; ++mt) {
      v16bf af = frag_a(As[buf], 32, wm + mt * 16, 0, lane);
#pragma unroll
      for (int nt = 0; nt < 2; ++nt) acc[mt][nt] = wmma_bf16(af, bfr[nt], acc[mt][nt]);
    }
  };

  const int nk = Kdim >> 5;
  issue(0, 0);
  commit(0);
  WAIT_ASYNC();
  __syncthreads();
  for (int kb = 0; kb < nk; ++kb) {
    const int cur = kb & 1;
    if (kb + 1 < nk) issue((kb + 1) << 5, cur ^ 1);   // overlaps with WMMAs
    compute(cur);
    if (kb + 1 < nk) {
      commit(cur ^ 1);
      WAIT_ASYNC();
      __syncthreads();
    }
  }

  const int ncol = lane & 15, hf = lane >> 4;
#pragma unroll
  for (int mt = 0; mt < 4; ++mt)
#pragma unroll
    for (int nt = 0; nt < 2; ++nt) {
      int gn = n0 + wn + nt * 16 + ncol;
      float bv = bias ? bias[gn] : 0.f;
#pragma unroll
      for (int r = 0; r < 8; ++r) {
        int gm = m0 + wm + mt * 16 + r + 8 * hf;
        float v = acc[mt][nt][r] + bv;
        if (resid) v += resid[(size_t)gm * Ndim + gn];
        if (relu)  v = v > 0.f ? v : 0.f;
        if (outF)  outF[(size_t)gm * Ndim + gn] = v;
        if (outB)  outB[(size_t)gm * Ndim + gn] = f32_to_bf16(v);
      }
    }
}

// ---------------- flash-style causal attention ----------------
// block: 256 thr (8 waves) = 128 query rows of one (b,h); 64-key tiles streamed.
__global__ __launch_bounds__(256)
void k_attention(const uint16_t* __restrict__ qb, const uint16_t* __restrict__ kb,
                 const uint16_t* __restrict__ vb, uint16_t* __restrict__ ob) {
  __shared__ uint16_t Qs[128 * 72];       // [qrow][d], rows 16B-aligned
  __shared__ uint16_t Ks[64 * 72];        // [key][d]   (B-frag for S: n=key)
  __shared__ uint16_t Vt[64 * 66];        // [d][key]   (B-frag for O: n=d)
  __shared__ uint16_t Ps[8 * 16 * 66];    // per-wave P slab [16][64] (+pad)

  const int QT = T_ / 128;
  const int bh = blockIdx.x / QT, qt = blockIdx.x % QT;
  const int b  = bh / H_, h = bh % H_;
  const int q0 = qt * 128;
  const int tid = threadIdx.x, lane = tid & 31, wid = tid >> 5;
  const int wrow0 = wid * 16;
  const size_t rowbase = (size_t)(b * T_) * E_ + (size_t)h * HS_;

#pragma unroll
  for (int i = 0; i < 4; ++i) {                         // Q tile 128x64
    int li = tid + 256 * i, row = li >> 3, seg = li & 7;
    const uint16_t* src = qb + rowbase + (size_t)(q0 + row) * E_ + seg * 8;
#if HAVE_ASYNC
    ASYNC_COPY16(&Qs[row * 72 + seg * 8], src);
#else
    uint4 v = *(const uint4*)src;
    uint32_t* d = (uint32_t*)&Qs[(size_t)row * 72 + seg * 8];
    const uint32_t* s = (const uint32_t*)&v;
    d[0] = s[0]; d[1] = s[1]; d[2] = s[2]; d[3] = s[3];
#endif
  }

  v8f accO[4];
#pragma unroll
  for (int dt = 0; dt < 4; ++dt)
#pragma unroll
    for (int e = 0; e < 8; ++e) accO[dt][e] = 0.f;
  float mstat[8], lstat[8];
#pragma unroll
  for (int r = 0; r < 8; ++r) { mstat[r] = -1e30f; lstat[r] = 0.f; }

  const int ncol = lane & 15, hf = lane >> 4;
  const int nkt = 2 * qt + 2;                           // keys < q0+128
  for (int kt = 0; kt < nkt; ++kt) {
    const int kbase = kt * 64;
    __syncthreads();                                    // Ks/Vt reuse guard
#if HAVE_ASYNC
#pragma unroll
    for (int i = 0; i < 2; ++i) {                       // K tile 64x64 async
      int li = tid + 256 * i, row = li >> 3, seg = li & 7;
      ASYNC_COPY16(&Ks[row * 72 + seg * 8],
                   kb + rowbase + (size_t)(kbase + row) * E_ + seg * 8);
    }
#else
    {
      uint4 kr2[2];
#pragma unroll
      for (int i = 0; i < 2; ++i) {
        int li = tid + 256 * i, row = li >> 3, seg = li & 7;
        kr2[i] = *(const uint4*)(kb + rowbase + (size_t)(kbase + row) * E_ + seg * 8);
      }
#pragma unroll
      for (int i = 0; i < 2; ++i) {
        int li = tid + 256 * i, row = li >> 3, seg = li & 7;
        uint32_t* d = (uint32_t*)&Ks[(size_t)row * 72 + seg * 8];
        const uint32_t* s = (const uint32_t*)&kr2[i];
        d[0] = s[0]; d[1] = s[1]; d[2] = s[2]; d[3] = s[3];
      }
    }
#endif
    {
      uint4 vr[2];                                      // V tile, transpose via regs
#pragma unroll
      for (int i = 0; i < 2; ++i) {
        int li = tid + 256 * i, row = li >> 3, seg = li & 7;
        vr[i] = *(const uint4*)(vb + rowbase + (size_t)(kbase + row) * E_ + seg * 8);
      }
#pragma unroll
      for (int i = 0; i < 2; ++i) {
        int li = tid + 256 * i, row = li >> 3, seg = li & 7;
        const uint16_t* pv = (const uint16_t*)&vr[i];
#pragma unroll
        for (int e = 0; e < 8; ++e) Vt[(size_t)(seg * 8 + e) * 66 + row] = pv[e];
      }
    }
    WAIT_ASYNC();
    __syncthreads();

    v8f s4[4];                                          // S tile 16x64 per wave
#pragma unroll
    for (int t = 0; t < 4; ++t)
#pragma unroll
      for (int e = 0; e < 8; ++e) s4[t][e] = 0.f;
    for (int kk = 0; kk < 64; kk += 32) {
      v16bf af = frag_a(Qs, 72, wrow0, kk, lane);
#pragma unroll
      for (int t = 0; t < 4; ++t)
        s4[t] = wmma_bf16(af, frag_b(Ks, 72, t * 16, kk, lane), s4[t]);
    }

    float rmax[8];
#pragma unroll
    for (int r = 0; r < 8; ++r) rmax[r] = -1e30f;
#pragma unroll
    for (int t = 0; t < 4; ++t) {
      int col = kbase + t * 16 + ncol;
#pragma unroll
      for (int r = 0; r < 8; ++r) {
        int rowg = q0 + wrow0 + r + 8 * hf;
        float sv = s4[t][r] * 0.125f;                   // 1/sqrt(64)
        if (col > rowg) sv = -1e30f;                    // causal mask
        s4[t][r] = sv;
        rmax[r] = fmaxf(rmax[r], sv);
      }
    }
#pragma unroll
    for (int r = 0; r < 8; ++r)
#pragma unroll
      for (int o = 1; o < 16; o <<= 1) rmax[r] = fmaxf(rmax[r], __shfl_xor(rmax[r], o, 32));

    float alpha[8], rsum[8];
#pragma unroll
    for (int r = 0; r < 8; ++r) {
      float mn = fmaxf(mstat[r], rmax[r]);
      alpha[r] = __expf(mstat[r] - mn);
      mstat[r] = mn;
      rsum[r] = 0.f;
    }
    uint16_t* Pw = Ps + (size_t)wid * 16 * 66;
#pragma unroll
    for (int t = 0; t < 4; ++t)
#pragma unroll
      for (int r = 0; r < 8; ++r) {
        float p = __expf(s4[t][r] - mstat[r]);
        rsum[r] += p;
        Pw[(size_t)(r + 8 * hf) * 66 + t * 16 + ncol] = f32_to_bf16(p);
      }
#pragma unroll
    for (int r = 0; r < 8; ++r) {
#pragma unroll
      for (int o = 1; o < 16; o <<= 1) rsum[r] += __shfl_xor(rsum[r], o, 32);
      lstat[r] = lstat[r] * alpha[r] + rsum[r];
    }
#pragma unroll
    for (int dt = 0; dt < 4; ++dt)
#pragma unroll
      for (int r = 0; r < 8; ++r) accO[dt][r] *= alpha[r];

    for (int kk = 0; kk < 64; kk += 32) {               // O += P @ V (wave-local)
      v16bf af = frag_a(Pw, 66, 0, kk, lane);
#pragma unroll
      for (int dt = 0; dt < 4; ++dt)
        accO[dt] = wmma_bf16(af, frag_b(Vt, 66, dt * 16, kk, lane), accO[dt]);
    }
  }

#pragma unroll
  for (int dt = 0; dt < 4; ++dt)
#pragma unroll
    for (int r = 0; r < 8; ++r) {
      float v = accO[dt][r] / lstat[r];
      int rowg = q0 + wrow0 + r + 8 * hf;
      ob[rowbase + (size_t)rowg * E_ + dt * 16 + ncol] = f32_to_bf16(v);
    }
}

// ---------------- loss ----------------
__global__ void k_zero(float* p) { p[0] = 0.f; }

__global__ __launch_bounds__(256) void k_loss(const float* __restrict__ logits,
                                              const int* __restrict__ tgt,
                                              float* __restrict__ accum) {
  int row = blockIdx.x;
  const float* lr = logits + (size_t)row * V_;
  __shared__ float red[256];
  float mx = -1e30f;
  for (int i = threadIdx.x; i < V_; i += 256) mx = fmaxf(mx, lr[i]);
  red[threadIdx.x] = mx; __syncthreads();
  for (int o = 128; o > 0; o >>= 1) { if (threadIdx.x < o) red[threadIdx.x] = fmaxf(red[threadIdx.x], red[threadIdx.x + o]); __syncthreads(); }
  mx = red[0]; __syncthreads();
  float s = 0.f;
  for (int i = threadIdx.x; i < V_; i += 256) s += __expf(lr[i] - mx);
  red[threadIdx.x] = s; __syncthreads();
  for (int o = 128; o > 0; o >>= 1) { if (threadIdx.x < o) red[threadIdx.x] += red[threadIdx.x + o]; __syncthreads(); }
  if (threadIdx.x == 0) atomicAdd(accum, logf(red[0]) + mx - lr[tgt[row]]);
}

__global__ void k_final(const float* accum, float* out) { out[0] = accum[0] / (float)M_; }

// ---------------- launch ----------------
extern "C" void kernel_launch(void* const* d_in, const int* in_sizes, int n_in,
                              void* d_out, int out_size, void* d_ws, size_t ws_size,
                              hipStream_t stream) {
  const int*   idx  = (const int*)d_in[0];
  const int*   tgt  = (const int*)d_in[1];
  const float* tok  = (const float*)d_in[2];
  const float* pos  = (const float*)d_in[3];
  const float* wq   = (const float*)d_in[4];
  const float* wk   = (const float*)d_in[5];
  const float* wv   = (const float*)d_in[6];
  const float* wo   = (const float*)d_in[7];
  const float* bo   = (const float*)d_in[8];
  const float* ln1s = (const float*)d_in[9];
  const float* ln1b = (const float*)d_in[10];
  const float* w1   = (const float*)d_in[11];
  const float* b1   = (const float*)d_in[12];
  const float* w2   = (const float*)d_in[13];
  const float* b2   = (const float*)d_in[14];
  const float* ln2s = (const float*)d_in[15];
  const float* ln2b = (const float*)d_in[16];
  const float* lnfs = (const float*)d_in[17];
  const float* lnfb = (const float*)d_in[18];
  const float* lmw  = (const float*)d_in[19];
  const float* lmb  = (const float*)d_in[20];

  char* ws = (char*)d_ws;
  size_t off = 0;
  auto take = [&](size_t bytes) { void* p = ws + off; off += (bytes + 255) & ~(size_t)255; return p; };

  float*    x      = (float*)   take((size_t)M_ * E_ * 4);
  uint16_t* hbf    = (uint16_t*)take((size_t)M_ * E_ * 2);
  uint16_t* qbf    = (uint16_t*)take((size_t)M_ * E_ * 2);
  uint16_t* kbf    = (uint16_t*)take((size_t)M_ * E_ * 2);
  uint16_t* vbf    = (uint16_t*)take((size_t)M_ * E_ * 2);
  uint16_t* attbf  = (uint16_t*)take((size_t)M_ * E_ * 2);
  uint16_t* hidbf  = (uint16_t*)take((size_t)M_ * 4 * E_ * 2);
  uint16_t* wq_bf  = (uint16_t*)take((size_t)L_ * E_ * E_ * 2);
  uint16_t* wk_bf  = (uint16_t*)take((size_t)L_ * E_ * E_ * 2);
  uint16_t* wv_bf  = (uint16_t*)take((size_t)L_ * E_ * E_ * 2);
  uint16_t* wo_bf  = (uint16_t*)take((size_t)L_ * E_ * E_ * 2);
  uint16_t* w1_bf  = (uint16_t*)take((size_t)L_ * E_ * 4 * E_ * 2);
  uint16_t* w2_bf  = (uint16_t*)take((size_t)L_ * 4 * E_ * E_ * 2);
  uint16_t* lmw_bf = (uint16_t*)take((size_t)E_ * V_ * 2);
  float*    lacc   = (float*)   take(256);

  k_zero<<<1, 1, 0, stream>>>(lacc);

  k_f32_to_bf16<<<4096, 256, 0, stream>>>(wq,  wq_bf,  (long)L_ * E_ * E_);
  k_f32_to_bf16<<<4096, 256, 0, stream>>>(wk,  wk_bf,  (long)L_ * E_ * E_);
  k_f32_to_bf16<<<4096, 256, 0, stream>>>(wv,  wv_bf,  (long)L_ * E_ * E_);
  k_f32_to_bf16<<<4096, 256, 0, stream>>>(wo,  wo_bf,  (long)L_ * E_ * E_);
  k_f32_to_bf16<<<8192, 256, 0, stream>>>(w1,  w1_bf,  (long)L_ * E_ * 4 * E_);
  k_f32_to_bf16<<<8192, 256, 0, stream>>>(w2,  w2_bf,  (long)L_ * 4 * E_ * E_);
  k_f32_to_bf16<<<8192, 256, 0, stream>>>(lmw, lmw_bf, (long)E_ * V_);

  k_embed<<<(M_ * E_ + 255) / 256, 256, 0, stream>>>(idx, tok, pos, x);

  dim3 gE(E_ / 128, M_ / 128);
  dim3 gH(4 * E_ / 128, M_ / 128);
  for (int l = 0; l < L_; ++l) {
    size_t we = (size_t)l * E_ * E_;
    k_layernorm<<<M_, 256, 0, stream>>>(x, ln1s + (size_t)l * E_, ln1b + (size_t)l * E_, hbf);
    k_gemm<<<gE, 256, 0, stream>>>(hbf, wq_bf + we, nullptr, nullptr, nullptr, qbf, M_, E_, E_, 0);
    k_gemm<<<gE, 256, 0, stream>>>(hbf, wk_bf + we, nullptr, nullptr, nullptr, kbf, M_, E_, E_, 0);
    k_gemm<<<gE, 256, 0, stream>>>(hbf, wv_bf + we, nullptr, nullptr, nullptr, vbf, M_, E_, E_, 0);
    k_attention<<<B_ * H_ * (T_ / 128), 256, 0, stream>>>(qbf, kbf, vbf, attbf);
    k_gemm<<<gE, 256, 0, stream>>>(attbf, wo_bf + we, bo + (size_t)l * E_, x, x, nullptr, M_, E_, E_, 0);
    k_layernorm<<<M_, 256, 0, stream>>>(x, ln2s + (size_t)l * E_, ln2b + (size_t)l * E_, hbf);
    k_gemm<<<gH, 256, 0, stream>>>(hbf, w1_bf + (size_t)l * E_ * 4 * E_, b1 + (size_t)l * 4 * E_,
                                   nullptr, nullptr, hidbf, M_, 4 * E_, E_, 1);
    k_gemm<<<gE, 256, 0, stream>>>(hidbf, w2_bf + (size_t)l * 4 * E_ * E_, b2 + (size_t)l * E_,
                                   x, x, nullptr, M_, E_, 4 * E_, 0);
  }

  k_layernorm<<<M_, 256, 0, stream>>>(x, lnfs, lnfb, hbf);
  float* logits = (float*)d_out;
  dim3 gL(V_ / 128, M_ / 128);
  k_gemm<<<gL, 256, 0, stream>>>(hbf, lmw_bf, lmb, nullptr, logits, nullptr, M_, V_, E_, 0);

  k_loss<<<M_, 256, 0, stream>>>(logits, tgt, lacc);
  k_final<<<1, 1, 0, stream>>>(lacc, logits + (size_t)M_ * V_);
}
